// ImageTextMambaModel_51264729645605
// MI455X (gfx1250) — matrix operand order repaired and checked by reading, not compiled, not run on hardware
//
#include <hip/hip_runtime.h>

// ---------------------------------------------------------------------------
// ImageTextMamba on MI455X (gfx1250, wave32).
//  - Big GEMMs: v_wmma_f32_16x16x32_f16, double-buffered LDS fed by
//    GLOBAL_LOAD_ASYNC_TO_LDS_B128 (ASYNCcnt), fragments via ds_load_b128.
//  - Producers (LN / scan / gather) emit f16 activations; weights are
//    transposed+converted to f16 N-major once per call.
//  - Scan/LN/gate/head: VALU f32 (latency/bandwidth trivial).
// ---------------------------------------------------------------------------

typedef _Float16 v16h __attribute__((ext_vector_type(16)));
typedef _Float16 v8h  __attribute__((ext_vector_type(8)));
typedef _Float16 v4h  __attribute__((ext_vector_type(4)));
typedef float    v8f  __attribute__((ext_vector_type(8)));
typedef float    v4f  __attribute__((ext_vector_type(4)));

#define D_MODEL 512
#define D_INNER 1024
#define D_STATE 16
#define N_LAYERS 4
#define BATCH 32
#define L_TXT 128
#define SEQ 129
#define ROWS (BATCH * SEQ)   // 4128

// LDS tile row stride: 40 halves = 80B (16B aligned, bank-rotating)
#define LDS_STRIDE 40

// ---- CDNA5 async global->LDS copy (16B per lane), ASYNCcnt-tracked --------
__device__ __forceinline__ void async_copy_b128(unsigned lds_off,
                                                unsigned long long gaddr)
{
    asm volatile("global_load_async_to_lds_b128 %0, %1, off"
                 :: "v"(lds_off), "v"(gaddr) : "memory");
}

#if __has_builtin(__builtin_amdgcn_s_wait_asynccnt)
#define WAIT_ASYNC(n) __builtin_amdgcn_s_wait_asynccnt(n)
#else
#define WAIT_ASYNC(n) asm volatile("s_wait_asynccnt %0" :: "i"(n) : "memory")
#endif

// ---------------------------------------------------------------------------
// WMMA GEMM: C[M,N](f32) = A[M,K](f16, row-major) @ Bt[N,K](f16, N-major)
// Optional bias[N]; optional residual (may alias C).
// Requires K % 32 == 0, N % 64 == 0; M handled by clamping (tail C rows are
// computed on duplicated A rows but never stored).
// Block = 128 threads = 4 waves; block tile 64x64; wave tile 32x32.
// ---------------------------------------------------------------------------
__launch_bounds__(128)
__global__ void gemm_wmma_f16(const _Float16* __restrict__ A,
                              const _Float16* __restrict__ Bt,
                              float* __restrict__ C,
                              const float* __restrict__ bias,
                              const float* __restrict__ resid,
                              int M, int N, int K)
{
    __shared__ __align__(16) _Float16 As[2][64 * LDS_STRIDE];
    __shared__ __align__(16) _Float16 Bs[2][64 * LDS_STRIDE];   // Bs[n][k]

    const int tid  = threadIdx.x;
    const int lane = tid & 31;
    const int wave = tid >> 5;            // 0..3
    const int wm   = (wave >> 1) * 32;
    const int wn   = (wave & 1)  * 32;

    const int tileM = blockIdx.y * 64;
    const int tileN = blockIdx.x * 64;

    const int halfSel = lane >> 4;
    const int l15     = lane & 15;
    const int a_kb    = halfSel * 8;      // A fragment K-base (halves)
    const int b_kb    = halfSel * 8;      // B fragment K-base (halves)

    // per-thread staging coordinates: 256 x 16B chunks per tile, 2 per thread
    const int r0  = tid >> 1;                 // j=0: rows 0..63 (chunk pairs)
    // chunk i = tid + 128*j ; r = i>>2 ; c8 = (i&3)*8
    v8f acc[2][2] = {};

    const int nk = K >> 5;

    // ---- stage K-step 0 into buffer 0 ----
    {
        unsigned asb = (unsigned)(unsigned long long)&As[0][0];
        unsigned bsb = (unsigned)(unsigned long long)&Bs[0][0];
        #pragma unroll
        for (int j = 0; j < 2; ++j) {
            int i  = tid + j * 128;
            int r  = i >> 2;
            int c8 = (i & 3) << 3;
            int gr = tileM + r; if (gr >= M) gr = M - 1;
            async_copy_b128(asb + (unsigned)(r * LDS_STRIDE + c8) * 2,
                            (unsigned long long)(A + (size_t)gr * K + c8));
            int nr = tileN + r;
            async_copy_b128(bsb + (unsigned)(r * LDS_STRIDE + c8) * 2,
                            (unsigned long long)(Bt + (size_t)nr * K + c8));
        }
    }

    for (int kk = 0; kk < nk; ++kk) {
        const int cur = kk & 1;
        if (kk + 1 < nk) {
            // prefetch next K-step into other buffer (protected by the
            // end-of-compute barrier of iteration kk-1)
            const int nxt = cur ^ 1;
            const int k0  = (kk + 1) << 5;
            unsigned asb = (unsigned)(unsigned long long)&As[nxt][0];
            unsigned bsb = (unsigned)(unsigned long long)&Bs[nxt][0];
            #pragma unroll
            for (int j = 0; j < 2; ++j) {
                int i  = tid + j * 128;
                int r  = i >> 2;
                int c8 = (i & 3) << 3;
                int gr = tileM + r; if (gr >= M) gr = M - 1;
                async_copy_b128(asb + (unsigned)(r * LDS_STRIDE + c8) * 2,
                                (unsigned long long)(A + (size_t)gr * K + k0 + c8));
                int nr = tileN + r;
                async_copy_b128(bsb + (unsigned)(r * LDS_STRIDE + c8) * 2,
                                (unsigned long long)(Bt + (size_t)nr * K + k0 + c8));
            }
            WAIT_ASYNC(4);   // in-order: first 4 (stage kk) have landed
        } else {
            WAIT_ASYNC(0);
        }
        __syncthreads();

        // ---- fragments (ds_load_b128 pairs; ISA 16-bit A/B layouts) ----
        const _Float16* Ab = &As[cur][0];
        const _Float16* Bb = &Bs[cur][0];
        v16h afrag[2], bfrag[2];
        #pragma unroll
        for (int s = 0; s < 2; ++s) {
            int row = wm + s * 16 + l15;
            v8h alo = *(const v8h*)(Ab + row * LDS_STRIDE + a_kb);
            v8h ahi = *(const v8h*)(Ab + row * LDS_STRIDE + 16 + a_kb);
            afrag[s] = __builtin_shufflevector(alo, ahi,
                0,1,2,3,4,5,6,7,8,9,10,11,12,13,14,15);
            int nrow = wn + s * 16 + l15;
            v8h blo = *(const v8h*)(Bb + nrow * LDS_STRIDE + b_kb);
            v8h bhi = *(const v8h*)(Bb + nrow * LDS_STRIDE + 16 + b_kb);
            bfrag[s] = __builtin_shufflevector(blo, bhi,
                0,1,2,3,4,5,6,7,8,9,10,11,12,13,14,15);
        }

        #pragma unroll
        for (int sm = 0; sm < 2; ++sm)
            #pragma unroll
            for (int sn = 0; sn < 2; ++sn)
                acc[sm][sn] = __builtin_amdgcn_wmma_f32_16x16x32_f16(
                    false, afrag[sm], false, bfrag[sn],
                    (short)0, acc[sm][sn], false, false);
        __syncthreads();
    }
    (void)r0;

    // ---- epilogue: lanes0-15 -> M=r, lanes16-31 -> M=8+r; N = l15 ----
    #pragma unroll
    for (int sm = 0; sm < 2; ++sm) {
        #pragma unroll
        for (int sn = 0; sn < 2; ++sn) {
            int col = tileN + wn + sn * 16 + l15;
            float bv = bias ? bias[col] : 0.0f;
            #pragma unroll
            for (int r = 0; r < 8; ++r) {
                int row = tileM + wm + sm * 16 + halfSel * 8 + r;
                if (row < M) {
                    float v = acc[sm][sn][r] + bv;
                    if (resid) v += resid[(size_t)row * N + col];
                    C[(size_t)row * N + col] = v;
                }
            }
        }
    }
}

// ---------------------------------------------------------------------------
// Weight prep: Wt[n][k] = (f16) W[k][n]
// ---------------------------------------------------------------------------
__global__ void transpose_f16_kernel(const float* __restrict__ W,
                                     _Float16* __restrict__ Wt, int K, int N)
{
    int idx = blockIdx.x * blockDim.x + threadIdx.x;
    if (idx >= K * N) return;
    int k = idx / N;
    int n = idx - k * N;
    Wt[(size_t)n * K + k] = (_Float16)W[idx];
}

// f32 -> f16 (contiguous), n4 = count/4
__global__ void convert_f16_kernel(const float* __restrict__ in,
                                   _Float16* __restrict__ out, int n4)
{
    int idx = blockIdx.x * blockDim.x + threadIdx.x;
    if (idx >= n4) return;
    v4f v = *(const v4f*)(in + (size_t)idx * 4);
    v4h o = { (_Float16)v.x, (_Float16)v.y, (_Float16)v.z, (_Float16)v.w };
    *(v4h*)(out + (size_t)idx * 4) = o;
}

// ---------------------------------------------------------------------------
// Embedding gather -> f16
// ---------------------------------------------------------------------------
__global__ void gather_kernel(const int* __restrict__ ids,
                              const float* __restrict__ table,
                              _Float16* __restrict__ emb)
{
    int idx = blockIdx.x * blockDim.x + threadIdx.x;   // 4096*128 float4 chunks
    int row = idx >> 7;
    int c   = (idx & 127) << 2;
    int id  = ids[row];
    v4f v = *(const v4f*)(table + (size_t)id * D_MODEL + c);
    v4h o = { (_Float16)v.x, (_Float16)v.y, (_Float16)v.z, (_Float16)v.w };
    *(v4h*)(emb + (size_t)row * D_MODEL + c) = o;
}

// ---------------------------------------------------------------------------
// Assemble x = concat(img_proj, txt_proj*mask) + pos_emb   (f32 stream)
// ---------------------------------------------------------------------------
__global__ void assemble_kernel(const float* __restrict__ tmp_img,
                                const float* __restrict__ tmp_txt,
                                const float* __restrict__ mask,
                                const float* __restrict__ pos,
                                float* __restrict__ x)
{
    int idx = blockIdx.x * blockDim.x + threadIdx.x;   // ROWS*128 float4 chunks
    if (idx >= ROWS * 128) return;
    int row = idx >> 7;
    int c   = (idx & 127) << 2;
    int b   = row / SEQ;
    int t   = row - b * SEQ;
    v4f p = *(const v4f*)(pos + (size_t)t * D_MODEL + c);
    v4f v;
    if (t == 0) {
        v = *(const v4f*)(tmp_img + (size_t)b * D_MODEL + c);
    } else {
        float m = mask[b * L_TXT + (t - 1)];
        v4f tv = *(const v4f*)(tmp_txt + ((size_t)b * L_TXT + (t - 1)) * D_MODEL + c);
        v = tv * m;
    }
    v = v + p;
    *(v4f*)(x + (size_t)row * D_MODEL + c) = v;
}

// ---------------------------------------------------------------------------
// LayerNorm over 512, writes f16 h (next GEMM's A operand)
// ---------------------------------------------------------------------------
__launch_bounds__(128)
__global__ void ln_kernel(const float* __restrict__ x, _Float16* __restrict__ h,
                          const float* __restrict__ g, const float* __restrict__ b)
{
    __shared__ float red[128];
    int row = blockIdx.x;
    int tid = threadIdx.x;
    const float* xr = x + (size_t)row * D_MODEL;

    v4f v = *(const v4f*)(xr + tid * 4);
    red[tid] = v.x + v.y + v.z + v.w;
    __syncthreads();
    for (int off = 64; off; off >>= 1) {
        if (tid < off) red[tid] += red[tid + off];
        __syncthreads();
    }
    float mu = red[0] * (1.0f / (float)D_MODEL);
    __syncthreads();

    v4f d = { v.x - mu, v.y - mu, v.z - mu, v.w - mu };
    red[tid] = d.x * d.x + d.y * d.y + d.z * d.z + d.w * d.w;
    __syncthreads();
    for (int off = 64; off; off >>= 1) {
        if (tid < off) red[tid] += red[tid + off];
        __syncthreads();
    }
    float rstd = rsqrtf(red[0] * (1.0f / (float)D_MODEL) + 1e-5f);

    v4f gg = *(const v4f*)(g + tid * 4);
    v4f bb = *(const v4f*)(b + tid * 4);
    v4h o  = { (_Float16)(d.x * rstd * gg.x + bb.x),
               (_Float16)(d.y * rstd * gg.y + bb.y),
               (_Float16)(d.z * rstd * gg.z + bb.z),
               (_Float16)(d.w * rstd * gg.w + bb.w) };
    *(v4h*)(h + (size_t)row * D_MODEL + tid * 4) = o;
}

// ---------------------------------------------------------------------------
// gate[row][s] = sigmoid( sum_n u[row][n] * A[n][s] )
// One 128-thread block per row: 8 n-groups x 16 states, LDS reduce.
// ---------------------------------------------------------------------------
__launch_bounds__(128)
__global__ void gate_kernel(const float* __restrict__ u,
                            const float* __restrict__ Am,
                            float* __restrict__ gate)
{
    __shared__ float red[128];
    int row = blockIdx.x;
    int tid = threadIdx.x;
    int s   = tid & 15;
    int g   = tid >> 4;                    // 0..7
    const float* ur = u + (size_t)row * D_INNER + g * 128;
    const float* Ap = Am + (size_t)(g * 128) * D_STATE + s;
    float acc = 0.0f;
    #pragma unroll 4
    for (int n = 0; n < 128; ++n)
        acc += ur[n] * Ap[n * D_STATE];
    red[tid] = acc;
    __syncthreads();
    if (tid < 64) red[tid] += red[tid + 64];
    __syncthreads();
    if (tid < 32) red[tid] += red[tid + 32];
    __syncthreads();
    if (tid < 16) {
        float v = red[tid] + red[tid + 16];
        gate[(size_t)row * D_STATE + tid] = 1.0f / (1.0f + __expf(-v));
    }
}

// ---------------------------------------------------------------------------
// Selective scan: one thread per (batch, channel); writes f16 ys
// ---------------------------------------------------------------------------
__launch_bounds__(256)
__global__ void scan_kernel(const float* __restrict__ u,
                            const float* __restrict__ gate,
                            const float* __restrict__ Bm,
                            const float* __restrict__ Cm,
                            _Float16* __restrict__ ys)
{
    int idx = blockIdx.x * blockDim.x + threadIdx.x;   // BATCH * D_INNER
    int b = idx >> 10;
    int n = idx & 1023;

    const float* Bp = Bm + n * D_STATE;
    const float* Cp = Cm + n * D_STATE;
    v4f B0 = *(const v4f*)(Bp + 0),  B1 = *(const v4f*)(Bp + 4);
    v4f B2 = *(const v4f*)(Bp + 8),  B3 = *(const v4f*)(Bp + 12);
    v4f C0 = *(const v4f*)(Cp + 0),  C1 = *(const v4f*)(Cp + 4);
    v4f C2 = *(const v4f*)(Cp + 8),  C3 = *(const v4f*)(Cp + 12);

    v4f h0 = {}, h1 = {}, h2 = {}, h3 = {};
    for (int t = 0; t < SEQ; ++t) {
        size_t r = (size_t)(b * SEQ + t);
        const float* gp = gate + r * D_STATE;
        v4f g0 = *(const v4f*)(gp + 0),  g1 = *(const v4f*)(gp + 4);
        v4f g2 = *(const v4f*)(gp + 8),  g3 = *(const v4f*)(gp + 12);
        float ut = u[r * D_INNER + n];
        h0 = h0 * g0 + B0 * ut;
        h1 = h1 * g1 + B1 * ut;
        h2 = h2 * g2 + B2 * ut;
        h3 = h3 * g3 + B3 * ut;
        v4f y4 = h0 * C0 + h1 * C1 + h2 * C2 + h3 * C3;
        ys[r * D_INNER + n] = (_Float16)(y4.x + y4.y + y4.z + y4.w);
    }
}

// ---------------------------------------------------------------------------
// Head: score[b] = x[b,0,:] . W_head + b_head
// ---------------------------------------------------------------------------
__launch_bounds__(128)
__global__ void head_kernel(const float* __restrict__ x,
                            const float* __restrict__ Wh,
                            const float* __restrict__ bh,
                            float* __restrict__ out)
{
    __shared__ float red[128];
    int b   = blockIdx.x;
    int tid = threadIdx.x;
    const float* xr = x + (size_t)b * SEQ * D_MODEL;
    v4f xv = *(const v4f*)(xr + tid * 4);
    v4f wv = *(const v4f*)(Wh + tid * 4);
    red[tid] = xv.x * wv.x + xv.y * wv.y + xv.z * wv.z + xv.w * wv.w;
    __syncthreads();
    for (int off = 64; off; off >>= 1) {
        if (tid < off) red[tid] += red[tid + off];
        __syncthreads();
    }
    if (tid == 0) out[b] = red[0] + bh[0];
}

// ---------------------------------------------------------------------------
extern "C" void kernel_launch(void* const* d_in, const int* in_sizes, int n_in,
                              void* d_out, int out_size, void* d_ws, size_t ws_size,
                              hipStream_t stream)
{
    const float* image_features = (const float*)d_in[0];
    const int*   input_ids      = (const int*)  d_in[1];
    const float* attn_mask      = (const float*)d_in[2];
    const float* emb_table      = (const float*)d_in[3];
    const float* W_img          = (const float*)d_in[4];
    const float* b_img          = (const float*)d_in[5];
    const float* W_txt          = (const float*)d_in[6];
    const float* b_txt          = (const float*)d_in[7];
    const float* pos_emb        = (const float*)d_in[8];
    const float* W_head         = (const float*)d_in[9];
    const float* b_head         = (const float*)d_in[10];
    float* score = (float*)d_out;

    // ---- workspace layout ----
    float* wsf     = (float*)d_ws;
    float* tmp_txt = wsf;                                   // 4096*512
    float* tmp_img = tmp_txt + (size_t)4096 * 512;          // 32*512
    float* x       = tmp_img + (size_t)32 * 512;            // ROWS*512
    float* u       = x       + (size_t)ROWS * 512;          // ROWS*1024
    float* gate    = u       + (size_t)ROWS * 1024;         // ROWS*16
    _Float16* wsh  = (_Float16*)(gate + (size_t)ROWS * 16);
    _Float16* emb_h  = wsh;                                 // 4096*512
    _Float16* img_h  = emb_h  + (size_t)4096 * 512;         // 32*2048
    _Float16* Wt_txt = img_h  + (size_t)32 * 2048;          // 512*512
    _Float16* Wt_img = Wt_txt + (size_t)512 * 512;          // 512*2048
    _Float16* h_h    = Wt_img + (size_t)512 * 2048;         // ROWS*512
    _Float16* ys_h   = h_h    + (size_t)ROWS * 512;         // ROWS*1024
    _Float16* Wt_in  = ys_h   + (size_t)ROWS * 1024;        // 4 * 1024*512
    _Float16* Wt_out = Wt_in  + (size_t)4 * 1024 * 512;     // 4 * 512*1024

    // ---- weight / input prep (f16, B transposed to N-major) ----
    transpose_f16_kernel<<<(512 * 512 + 255) / 256, 256, 0, stream>>>(
        W_txt, Wt_txt, 512, 512);
    transpose_f16_kernel<<<(2048 * 512 + 255) / 256, 256, 0, stream>>>(
        W_img, Wt_img, 2048, 512);
    for (int l = 0; l < N_LAYERS; ++l) {
        const float* Win  = (const float*)d_in[11 + l * 9 + 2];
        const float* Wout = (const float*)d_in[11 + l * 9 + 4];
        transpose_f16_kernel<<<(512 * 1024 + 255) / 256, 256, 0, stream>>>(
            Win, Wt_in + (size_t)l * 1024 * 512, 512, 1024);
        transpose_f16_kernel<<<(1024 * 512 + 255) / 256, 256, 0, stream>>>(
            Wout, Wt_out + (size_t)l * 512 * 1024, 1024, 512);
    }
    convert_f16_kernel<<<(32 * 2048 / 4 + 255) / 256, 256, 0, stream>>>(
        image_features, img_h, 32 * 2048 / 4);

    // ---- embeddings ----
    gather_kernel<<<(4096 * 128) / 256, 256, 0, stream>>>(input_ids, emb_table, emb_h);
    gemm_wmma_f16<<<dim3(512 / 64, 4096 / 64), 128, 0, stream>>>(
        emb_h, Wt_txt, tmp_txt, b_txt, nullptr, 4096, 512, 512);
    gemm_wmma_f16<<<dim3(512 / 64, 1), 128, 0, stream>>>(
        img_h, Wt_img, tmp_img, b_img, nullptr, 32, 512, 2048);
    assemble_kernel<<<(ROWS * 128 + 255) / 256, 256, 0, stream>>>(
        tmp_img, tmp_txt, attn_mask, pos_emb, x);

    // ---- mamba layers ----
    for (int l = 0; l < N_LAYERS; ++l) {
        const float* ln_g = (const float*)d_in[11 + l * 9 + 0];
        const float* ln_b = (const float*)d_in[11 + l * 9 + 1];
        const float* bin  = (const float*)d_in[11 + l * 9 + 3];
        const float* bout = (const float*)d_in[11 + l * 9 + 5];
        const float* Amat = (const float*)d_in[11 + l * 9 + 6];
        const float* Bmat = (const float*)d_in[11 + l * 9 + 7];
        const float* Cmat = (const float*)d_in[11 + l * 9 + 8];

        ln_kernel<<<ROWS, 128, 0, stream>>>(x, h_h, ln_g, ln_b);
        gemm_wmma_f16<<<dim3(1024 / 64, (ROWS + 63) / 64), 128, 0, stream>>>(
            h_h, Wt_in + (size_t)l * 1024 * 512, u, bin, nullptr, ROWS, 1024, 512);
        gate_kernel<<<ROWS, 128, 0, stream>>>(u, Amat, gate);
        scan_kernel<<<(BATCH * D_INNER) / 256, 256, 0, stream>>>(
            u, gate, Bmat, Cmat, ys_h);
        gemm_wmma_f16<<<dim3(512 / 64, (ROWS + 63) / 64), 128, 0, stream>>>(
            ys_h, Wt_out + (size_t)l * 512 * 1024, x, bout, x, ROWS, 512, 1024);
    }

    // ---- head ----
    head_kernel<<<BATCH, 128, 0, stream>>>(x, W_head, b_head, score);
}
